// DeformablePatchEmbed_GELU_7352984011448
// MI455X (gfx1250) — matrix operand order repaired and checked by reading, not compiled, not run on hardware
//
#include <hip/hip_runtime.h>
#include <hip/hip_bf16.h>
#include <math.h>

// ---------------------------------------------------------------------------
// DeformablePatchEmbed + BN + GELU for MI455X (gfx1250, wave32, WMMA + TDM).
//
//   1) im2col  : x [64,3,224,224] -> P f16 [12544, 768]   (pure relayout)
//   2) cvt     : offset_w -> W1h f16 [512,768], dconv_w -> W2h f16 [768,768]
//   3) GEMM1   : off = P x W1h^T + bias      (f32 out, [12544,512])
//   4) sample  : bilinear deformable gather  -> Ah f16 [12544, 768]
//   5) GEMM2   : Y  = Ah x W2h^T             -> d_out (already [B,N,O] layout)
//   6) stats   : per-channel sum / sumsq over 12544 rows (atomics)
//   7) bn+gelu : in-place on d_out (exact GELU via erff)
//
// GEMM: f16 WMMA (16x16x32) with f32 accumulation. B panels are DMA'd into
// LDS by the Tensor Data Mover (double-buffered, TENSORcnt-tracked) so the
// WMMA stream overlaps the B-operand fetch with no VGPR round trip.
// ---------------------------------------------------------------------------

typedef __attribute__((ext_vector_type(16))) _Float16 v16h;
typedef __attribute__((ext_vector_type(8)))  _Float16 v8h;
typedef __attribute__((ext_vector_type(8)))  float    v8f;
typedef unsigned int u32x4 __attribute__((ext_vector_type(4)));
typedef int          i32x8 __attribute__((ext_vector_type(8)));
typedef int          i32x4 __attribute__((ext_vector_type(4)));

#define BB     64
#define CC     3
#define HH     224
#define WW     224
#define HO     14
#define WOW    14
#define PKK    256                 // 16x16 kernel points
#define KD     768                 // inner GEMM dim = C * 256
#define N1     512                 // offset channels (2K)
#define N2     768                 // embed dim
#define MROWS  (BB * HO * WOW)     // 12544
#define PANEL_K 192                // K extent of one LDS-resident B panel
#define NPANEL  (KD / PANEL_K)     // 4

union V16U { v16h v; struct { v8h lo; v8h hi; } p; };
static __device__ inline v16h make_v16(v8h lo, v8h hi) {
  V16U u; u.p.lo = lo; u.p.hi = hi; return u.v;
}

#if __has_builtin(__builtin_amdgcn_tensor_load_to_lds)
#define USE_TDM 1
#endif

#ifdef USE_TDM
// Generic pointers to LDS carry the LDS byte offset in their low 32 bits.
static __device__ inline unsigned lds_byte_off(const void* p) {
  return (unsigned)(unsigned long long)p;
}

// 2-D TDM load: tile_cols rows of tile_k f16 elements, row pitch
// row_stride_elems in global memory, packed [col][k] into LDS.
static __device__ inline void tdm_load_b_panel(unsigned lds_off,
                                               const void* global_tile,
                                               unsigned tile_k,
                                               unsigned tile_cols,
                                               unsigned row_stride_elems) {
  unsigned long long ga = (unsigned long long)global_tile;
  u32x4 g0;
  g0[0] = 1u;                                            // count=1 (valid D#)
  g0[1] = lds_off;                                       // lds_addr (bytes)
  g0[2] = (unsigned)ga;                                  // global_addr[31:0]
  g0[3] = (unsigned)((ga >> 32) & 0x01FFFFFFu) | (2u << 30); // [56:32] | type=2
  i32x8 g1;
  g1[0] = (int)(1u << 16);                               // data_size=1 -> 2B
  g1[1] = (int)((row_stride_elems & 0xFFFFu) << 16);     // tensor_dim0 lo16
  g1[2] = (int)(((row_stride_elems >> 16) & 0xFFFFu) |
                ((tile_cols & 0xFFFFu) << 16));          // dim0 hi | dim1 lo
  g1[3] = (int)(((tile_cols >> 16) & 0xFFFFu) |
                ((tile_k & 0xFFFFu) << 16));             // dim1 hi | tile_dim0
  g1[4] = (int)(tile_cols & 0xFFFFu);                    // tile_dim1, tile_dim2=0
  g1[5] = (int)row_stride_elems;                         // dim0_stride lo32
  g1[6] = 0;                                             // dim0_stride hi, dim1_stride lo
  g1[7] = 0;
  i32x4 gz4 = {0, 0, 0, 0};                              // groups 2/3: unused (2-D)
  i32x8 gz8 = {0, 0, 0, 0, 0, 0, 0, 0};                  // trailing group: unused
  __builtin_amdgcn_tensor_load_to_lds(g0, g1, gz4, gz4, gz8, 0);
}
#endif

// ------------------------- 1) im2col (f32 -> f16) --------------------------
__global__ __launch_bounds__(256) void k_im2col(const float* __restrict__ x,
                                                _Float16* __restrict__ P) {
  size_t idx = (size_t)blockIdx.x * 256 + threadIdx.x;
  if (idx >= (size_t)MROWS * KD) return;
  int col = (int)(idx % KD);
  size_t row = idx / KD;
  int b  = (int)(row / (HO * WOW));
  int hw = (int)(row % (HO * WOW));
  int ho = hw / WOW, wo = hw % WOW;
  int c = col >> 8, rem = col & 255;
  int i = rem >> 4, j = rem & 15;
  float v = x[(((size_t)b * CC + c) * HH + (ho * 16 + i)) * WW + (wo * 16 + j)];
  P[idx] = (_Float16)v;
}

// ------------------------- 2) weight convert -------------------------------
__global__ __launch_bounds__(256) void k_cvt(const float* __restrict__ src,
                                             _Float16* __restrict__ dst, int n) {
  int i = blockIdx.x * 256 + threadIdx.x;
  if (i < n) dst[i] = (_Float16)src[i];
}

__global__ __launch_bounds__(256) void k_zero(float* __restrict__ p, int n) {
  int i = blockIdx.x * 256 + threadIdx.x;
  if (i < n) p[i] = 0.0f;
}

// ------------------- 3/5) f16 WMMA GEMM: C = A * Bt^T (+bias) --------------
// A : [M, K] f16 row-major (M multiple of 256, K multiple of PANEL_K)
// Bt: [N, K] f16 row-major (N multiple of 64)
// Block: 256 threads = 8 waves; wave computes 32x64; block tile 256x64.
// B panels (64 cols x PANEL_K) are TDM-DMA'd into double-buffered LDS.
__global__ __launch_bounds__(256) void k_gemm_f16(
    const _Float16* __restrict__ A, const _Float16* __restrict__ Bt,
    const float* __restrict__ bias, float* __restrict__ C, int N, int K) {
  __shared__ alignas(32) _Float16 Bs[2][64 * PANEL_K];   // 2 x 24 KB

  const int lane = threadIdx.x & 31;
  const int wave = threadIdx.x >> 5;
  const int m = lane & 15;     // row / col within fragment
  const int h = lane >> 4;     // lane-half select
  const int colBase = blockIdx.x * 64;
  const int rowBase = blockIdx.y * 256 + wave * 32;

  v8f acc[2][4];
#pragma unroll
  for (int i = 0; i < 2; ++i)
#pragma unroll
    for (int j = 0; j < 4; ++j)
      acc[i][j] = (v8f){};

#ifdef USE_TDM
  const bool issuer = (threadIdx.x < 32);   // wave 0 drives the TDM
  if (issuer) {
    tdm_load_b_panel(lds_byte_off(&Bs[0][0]),
                     Bt + (size_t)colBase * K, PANEL_K, 64, (unsigned)K);
  }
#endif

  for (int p = 0; p < NPANEL; ++p) {
    const _Float16* bs = Bs[p & 1];

#ifdef USE_TDM
    if (issuer) {
      if (p + 1 < NPANEL) {
        tdm_load_b_panel(lds_byte_off(&Bs[(p + 1) & 1][0]),
                         Bt + (size_t)colBase * K + (p + 1) * PANEL_K,
                         PANEL_K, 64, (unsigned)K);
        __builtin_amdgcn_s_wait_tensorcnt(1);   // panel p landed
      } else {
        __builtin_amdgcn_s_wait_tensorcnt(0);   // final drain
      }
    }
#else
    // manual staging fallback: 64 x PANEL_K halfs, 256 threads x 6 v8h
    {
      _Float16* dst = (_Float16*)Bs[p & 1];
      for (int u = threadIdx.x; u < 64 * PANEL_K / 8; u += 256) {
        int r = u / (PANEL_K / 8);
        int s = (u % (PANEL_K / 8)) * 8;
        *(v8h*)(dst + r * PANEL_K + s) =
            *(const v8h*)(Bt + (size_t)(colBase + r) * K + p * PANEL_K + s);
      }
    }
#endif
    __syncthreads();   // panel p visible to all waves

#pragma unroll
    for (int s = 0; s < PANEL_K; s += 32) {
      const int k0 = p * PANEL_K + s;
      // A fragments: lane(m,h) holds row m, K = h*16..h*16+15
      v16h a[2];
#pragma unroll
      for (int i = 0; i < 2; ++i) {
        const _Float16* ap = A + (size_t)(rowBase + i * 16 + m) * K + k0 + h * 16;
        a[i] = make_v16(*(const v8h*)ap, *(const v8h*)(ap + 8));
      }
      // B fragments from LDS: lane(n=m,h) holds col n, K = h*16..h*16+15
#pragma unroll
      for (int j = 0; j < 4; ++j) {
        v16h b = *(const v16h*)(bs + (j * 16 + m) * PANEL_K + s + h * 16);
#pragma unroll
        for (int i = 0; i < 2; ++i)
          acc[i][j] = __builtin_amdgcn_wmma_f32_16x16x32_f16(
              false, a[i], false, b, (short)0, acc[i][j], false, false);
      }
    }
    __syncthreads();   // all waves done with panel p before its buffer reloads
  }

  // D layout: VGPR r -> M = h*8 + r, N = m
#pragma unroll
  for (int j = 0; j < 4; ++j) {
    const int col = colBase + j * 16 + m;
    const float bv = bias ? bias[col] : 0.0f;
#pragma unroll
    for (int i = 0; i < 2; ++i) {
#pragma unroll
      for (int r = 0; r < 8; ++r) {
        const int row = rowBase + i * 16 + h * 8 + r;
        C[(size_t)row * N + col] = acc[i][j][r] + bv;
      }
    }
  }
}

// ------------------- 4) deformable bilinear sampling -----------------------
static __device__ inline float samp(const float* __restrict__ plane,
                                    int y, int x) {
  bool ok = (y >= 0) & (y < HH) & (x >= 0) & (x < WW);
  int yc = min(max(y, 0), HH - 1);
  int xc = min(max(x, 0), WW - 1);
  return ok ? plane[(size_t)yc * WW + xc] : 0.0f;
}

__global__ __launch_bounds__(256) void k_sample(const float* __restrict__ x,
                                                const float* __restrict__ off,
                                                _Float16* __restrict__ Ah) {
  const int row = blockIdx.x;        // 0..12543 : (b, ho, wo)
  const int k   = threadIdx.x;       // 0..255   : kernel point
  const int b  = row / (HO * WOW);
  const int hw = row % (HO * WOW);
  const int ho = hw / WOW, wo = hw % WOW;

  const float dy = off[(size_t)row * N1 + 2 * k];
  const float dx = off[(size_t)row * N1 + 2 * k + 1];
  const int ki = k >> 4, kj = k & 15;

  const float py = (float)(ho * 16 + ki) + dy;
  const float px = (float)(wo * 16 + kj) + dx;
  const float y0f = floorf(py), x0f = floorf(px);
  const float wy1 = py - y0f, wx1 = px - x0f;
  const float wy0 = 1.0f - wy1, wx0 = 1.0f - wx1;
  const int y0 = (int)y0f, x0 = (int)x0f;
  const int y1 = y0 + 1, x1 = x0 + 1;
  const float w00 = wy0 * wx0, w01 = wy0 * wx1;
  const float w10 = wy1 * wx0, w11 = wy1 * wx1;

#pragma unroll
  for (int c = 0; c < CC; ++c) {
    const float* plane = x + ((size_t)b * CC + c) * HH * WW;
    float v = w00 * samp(plane, y0, x0) + w01 * samp(plane, y0, x1) +
              w10 * samp(plane, y1, x0) + w11 * samp(plane, y1, x1);
    Ah[(size_t)row * KD + c * PKK + k] = (_Float16)v;
  }
}

// ------------------- 6) per-channel sums / sumsq ---------------------------
__global__ __launch_bounds__(256) void k_colstats(const float* __restrict__ Y,
                                                  float* __restrict__ sums,
                                                  float* __restrict__ sumsq) {
  const int t = threadIdx.x;            // covers 256 consecutive channels
  const int r0 = blockIdx.x * 128;      // 128-row chunk
  float s[3] = {0.f, 0.f, 0.f}, q[3] = {0.f, 0.f, 0.f};
  for (int r = r0; r < r0 + 128; ++r) {
#pragma unroll
    for (int j = 0; j < 3; ++j) {
      float v = Y[(size_t)r * N2 + j * 256 + t];
      s[j] += v;
      q[j] += v * v;
    }
  }
#pragma unroll
  for (int j = 0; j < 3; ++j) {
    atomicAdd(&sums[j * 256 + t], s[j]);
    atomicAdd(&sumsq[j * 256 + t], q[j]);
  }
}

// ------------------- 7) BatchNorm + exact GELU in place --------------------
__global__ __launch_bounds__(256) void k_bn_gelu(float* __restrict__ Y,
                                                 const float* __restrict__ sums,
                                                 const float* __restrict__ sumsq,
                                                 const float* __restrict__ gamma,
                                                 const float* __restrict__ beta) {
  size_t idx = (size_t)blockIdx.x * 256 + threadIdx.x;
  if (idx >= (size_t)MROWS * N2) return;
  const int o = (int)(idx % N2);
  const float inv_n = 1.0f / (float)MROWS;
  const float mean = sums[o] * inv_n;
  const float var  = sumsq[o] * inv_n - mean * mean;
  float v = (Y[idx] - mean) * rsqrtf(var + 1e-5f) * gamma[o] + beta[o];
  Y[idx] = 0.5f * v * (1.0f + erff(v * 0.70710678118654752440f));
}

// ---------------------------------------------------------------------------
extern "C" void kernel_launch(void* const* d_in, const int* in_sizes, int n_in,
                              void* d_out, int out_size, void* d_ws,
                              size_t ws_size, hipStream_t stream) {
  const float* x        = (const float*)d_in[0];
  const float* offset_w = (const float*)d_in[1];
  const float* offset_b = (const float*)d_in[2];
  const float* dconv_w  = (const float*)d_in[3];
  const float* bn_gamma = (const float*)d_in[4];
  const float* bn_beta  = (const float*)d_in[5];
  float* Y = (float*)d_out;

  // workspace carve-out (256B aligned slices)
  size_t cur = 0;
  auto wsAlloc = [&](size_t bytes) -> void* {
    void* p = (char*)d_ws + cur;
    cur += (bytes + 255) & ~(size_t)255;
    return p;
  };
  _Float16* P     = (_Float16*)wsAlloc((size_t)MROWS * KD * sizeof(_Float16));
  _Float16* W1h   = (_Float16*)wsAlloc((size_t)N1 * KD * sizeof(_Float16));
  _Float16* W2h   = (_Float16*)wsAlloc((size_t)N2 * KD * sizeof(_Float16));
  float*    OFF   = (float*)wsAlloc((size_t)MROWS * N1 * sizeof(float));
  _Float16* Ah    = (_Float16*)wsAlloc((size_t)MROWS * KD * sizeof(_Float16));
  float*    sums  = (float*)wsAlloc((size_t)2 * N2 * sizeof(float));
  float*    sumsq = sums + N2;

  // 1) im2col + 2) weight converts + zero stats
  {
    int total = MROWS * KD;                          // 9,633,792
    k_im2col<<<dim3((total + 255) / 256), dim3(256), 0, stream>>>(x, P);
  }
  k_cvt<<<dim3((N1 * KD + 255) / 256), dim3(256), 0, stream>>>(offset_w, W1h, N1 * KD);
  k_cvt<<<dim3((N2 * KD + 255) / 256), dim3(256), 0, stream>>>(dconv_w, W2h, N2 * KD);
  k_zero<<<dim3((2 * N2 + 255) / 256), dim3(256), 0, stream>>>(sums, 2 * N2);

  // 3) offset GEMM: [12544,768] x [512,768]^T + bias -> OFF
  k_gemm_f16<<<dim3(N1 / 64, MROWS / 256), dim3(256), 0, stream>>>(
      P, W1h, offset_b, OFF, N1, KD);

  // 4) deformable sampling -> Ah
  k_sample<<<dim3(MROWS), dim3(256), 0, stream>>>(x, OFF, Ah);

  // 5) main GEMM: [12544,768] x [768,768]^T -> d_out (already [B,N,O] layout)
  k_gemm_f16<<<dim3(N2 / 64, MROWS / 256), dim3(256), 0, stream>>>(
      Ah, W2h, nullptr, Y, N2, KD);

  // 6) channel stats, 7) BN + GELU
  k_colstats<<<dim3(MROWS / 128), dim3(256), 0, stream>>>(Y, sums, sumsq);
  {
    int total = MROWS * N2;
    k_bn_gelu<<<dim3((total + 255) / 256), dim3(256), 0, stream>>>(
        Y, sums, sumsq, bn_gamma, bn_beta);
  }
}
